// SurfNetwork_59631325937890
// MI455X (gfx1250) — matrix operand order repaired and checked by reading, not compiled, not run on hardware
//
#include <hip/hip_runtime.h>

typedef __attribute__((ext_vector_type(16))) _Float16 v16h;
typedef __attribute__((ext_vector_type(8)))  _Float16 v8h;
typedef __attribute__((ext_vector_type(8)))  float    v8f;
typedef __attribute__((ext_vector_type(4)))  float    f32x4;

#define NSAMP 128
#define INDIR 16

__device__ __forceinline__ float bperm(int srcLane, float v) {
  return __builtin_bit_cast(float,
      __builtin_amdgcn_ds_bpermute(srcLane << 2, __builtin_bit_cast(int, v)));
}

__device__ __forceinline__ float fsigmoid(float x) {
  return 1.0f / (1.0f + __expf(-x));
}

// Load a 16x32 f16 A-matrix fragment from row-major LDS in contiguous k order (matching B):
// lanes 0-15: row m=l, K in {0..15}; lanes 16-31: row m=l-16, K in {16..31}
__device__ __forceinline__ v16h load_A(const _Float16* base, int l) {
  const int m   = l & 15;
  const int klo = (l < 16) ? 0 : 16;
  v8h lo = *(const v8h*)(base + m * 32 + klo);
  v8h hi = *(const v8h*)(base + m * 32 + klo + 8);
  v16h r;
#pragma unroll
  for (int i = 0; i < 8; ++i) { r[i] = lo[i]; r[i + 8] = hi[i]; }
  return r;
}

__global__ __launch_bounds__(256) void surfnet_kernel(
    const int* __restrict__ x, const float* __restrict__ dvec,
    const float* __restrict__ grid, const float* __restrict__ W0,
    const float* __restrict__ W1, float* __restrict__ out_sigma,
    float* __restrict__ out_rgb, int B)
{
  // Block-shared (read-only after barrier): padded, transposed weights [n][k], f16
  __shared__ _Float16 sW0[16 * 32];
  __shared__ _Float16 sW1[16 * 32];
  // Per-wave staging buffers: A0 (input) and A1 (hidden) tiles, 16x32 f16 each
  __shared__ _Float16 sA[8][16 * 32];
  __shared__ _Float16 sH[8][16 * 32];

  const int tid  = threadIdx.x;
  const int wave = tid >> 5;
  const int l    = tid & 31;
  const int n    = l & 15;

  // Stage W0 (22x8) / W1 (8x3) transposed + zero-padded to 32x16 (stored [n][k])
  for (int i = tid; i < 16 * 32; i += 256) {
    int nn = i >> 5, kk = i & 31;
    sW0[i] = (_Float16)((kk < 22 && nn < 8) ? W0[kk * 8 + nn] : 0.0f);
    sW1[i] = (_Float16)((kk < 8  && nn < 3) ? W1[kk * 3 + nn] : 0.0f);
  }
  __syncthreads();

  const int b = blockIdx.x * 8 + wave;
  if (b >= B) return;

  // B fragments (f16 B 32x16, contiguous k order as in load_A): lane holds column n=l%16,
  // K = 0..15 (lanes 0-15) or 16..31 (lanes 16-31), pairs packed per VGPR.
  const int k0 = (l < 16) ? 0 : 16;
  const v16h fb0 = *(const v16h*)(&sW0[n * 32 + k0]);
  const v16h fb1 = *(const v16h*)(&sW1[n * 32 + k0]);

  _Float16* sAw = &sA[wave][0];
  _Float16* sHw = &sH[wave][0];

  const _Float16 Z = (_Float16)0.0f;
  const v8h zz = {Z, Z, Z, Z, Z, Z, Z, Z};

  // Per-ray one-time init: A cols 0..15 = d (f16), cols 22..31 = 0;
  // hidden-staging cols 8..31 = 0 (cols 0..7 refreshed every tile).
  if (l < 16) {
    const f32x4* dp = (const f32x4*)(dvec + (size_t)b * INDIR);
    const f32x4 q0 = dp[0], q1 = dp[1], q2 = dp[2], q3 = dp[3];
    v8h h0, h1;
    h0[0] = (_Float16)q0.x; h0[1] = (_Float16)q0.y;
    h0[2] = (_Float16)q0.z; h0[3] = (_Float16)q0.w;
    h0[4] = (_Float16)q1.x; h0[5] = (_Float16)q1.y;
    h0[6] = (_Float16)q1.z; h0[7] = (_Float16)q1.w;
    h1[0] = (_Float16)q2.x; h1[1] = (_Float16)q2.y;
    h1[2] = (_Float16)q2.z; h1[3] = (_Float16)q2.w;
    h1[4] = (_Float16)q3.x; h1[5] = (_Float16)q3.y;
    h1[6] = (_Float16)q3.z; h1[7] = (_Float16)q3.w;
    *(v8h*)(&sAw[l * 32 + 0]) = h0;
    *(v8h*)(&sAw[l * 32 + 8]) = h1;
    sAw[l * 32 + 22] = Z;
    sAw[l * 32 + 23] = Z;
    *(v8h*)(&sAw[l * 32 + 24]) = zz;
    *(v8h*)(&sHw[l * 32 + 8])  = zz;
    *(v8h*)(&sHw[l * 32 + 16]) = zz;
    *(v8h*)(&sHw[l * 32 + 24]) = zz;
  }

  float T_run = 1.0f;   // running transmittance across tiles
  float acc   = 0.0f;   // per-lane rgb partial (lanes with n<3)
  const int s  = l >> 1;  // sample within tile handled by this lane's gather
  const int jg = l & 1;   // which of the 2 grid rows

  // ---- software pipeline: prefetch tile 0's random 16B grid row ----
  const size_t xbase = ((size_t)b * NSAMP) * 2;
  {
    const int idx0 = x[xbase + (size_t)s * 2 + jg];
    // fall through into loop with fcur live
    f32x4 fcur = *(const f32x4*)(grid + (size_t)idx0 * 4);

    for (int t = 0; t < 8; ++t) {
      // prefetch next tile's gather while this tile computes (hides HBM latency)
      f32x4 fnxt = fcur;
      if (t < 7) {
        const int idxn = x[xbase + (size_t)((t + 1) * 16 + s) * 2 + jg];
        fnxt = *(const f32x4*)(grid + (size_t)idxn * 4);
      }

      // geo features -> A cols 16..21 of row s
      {
        _Float16* rp = &sAw[s * 32 + 16 + 3 * jg];
        rp[0] = (_Float16)fcur.y; rp[1] = (_Float16)fcur.z; rp[2] = (_Float16)fcur.w;
      }

      // ---- sigma: sigmoid(f0_j0 * f0_j1), then bring sample m to lane m ----
      const float sprod   = fcur.x * bperm(l ^ 1, fcur.x);
      const float sig_all = fsigmoid(sprod);
      const float sig     = bperm(n * 2, sig_all);  // lane m (and m+16) holds sample m

      // exclusive cumprod of (1 - sigma) over the 16 samples (wave scan)
      float incl = 1.0f - sig;
#pragma unroll
      for (int off = 1; off < 16; off <<= 1) {
        float up = bperm(l - off, incl);
        if (n >= off) incl *= up;
      }
      float exv = bperm(l - 1, incl);
      if (n == 0) exv = 1.0f;
      const float w = T_run * exv * sig;      // per-sample weight
      T_run *= bperm(15, incl);               // carry transmittance to next tile

      if (l < 16) out_sigma[(size_t)b * NSAMP + t * 16 + l] = sig;

      // ---- layer 0: [16x22] @ [22x8] via 16x16x32 f16 WMMA ----
      v8f c0 = {};
      const v16h a0 = load_A(sAw, l);
      v8f d0 = __builtin_amdgcn_wmma_f32_16x16x32_f16(
          false, a0, false, fb0, (short)0, c0, false, false);

      // relu -> restage as A of layer 1 (cols 0..7; rest pre-zeroed)
      if (n < 8) {
        const int mb = (l < 16) ? 0 : 8;
#pragma unroll
        for (int q = 0; q < 8; ++q) {
          float v = d0[q] > 0.0f ? d0[q] : 0.0f;
          sHw[(q + mb) * 32 + n] = (_Float16)v;
        }
      }

      // ---- layer 1: [16x8] @ [8x3] via WMMA ----
      v8f c1 = {};
      const v16h a1 = load_A(sHw, l);
      v8f d1 = __builtin_amdgcn_wmma_f32_16x16x32_f16(
          false, a1, false, fb1, (short)0, c1, false, false);

      // rgb accumulate: lane with column n<3 owns channel n for its 8 samples.
      // bperm executed by all lanes (disabled source lanes would read 0).
      const int wbase = (l < 16) ? 0 : 8;
#pragma unroll
      for (int q = 0; q < 8; ++q) {
        float wq = bperm(q + wbase, w);       // weight of sample q (+8 for hi half)
        if (n < 3) acc += fsigmoid(d1[q]) * wq;
      }

      fcur = fnxt;
    }
  }

  // combine the two half-wave partials (samples 0..7 on lanes 0..15, 8..15 on 16..31)
  acc += bperm(l ^ 16, acc);
  if (l < 3) out_rgb[(size_t)b * 3 + l] = acc;
}

extern "C" void kernel_launch(void* const* d_in, const int* in_sizes, int n_in,
                              void* d_out, int out_size, void* d_ws, size_t ws_size,
                              hipStream_t stream) {
  const int*   x  = (const int*)d_in[0];
  const float* dv = (const float*)d_in[1];
  const float* gw = (const float*)d_in[2];
  const float* W0 = (const float*)d_in[3];
  const float* W1 = (const float*)d_in[4];

  const int B = in_sizes[1] / INDIR;     // 32768
  float* out_sigma = (float*)d_out;
  float* out_rgb   = out_sigma + (size_t)B * NSAMP;

  dim3 grid((B + 7) / 8);
  surfnet_kernel<<<grid, 256, 0, stream>>>(x, dv, gw, W0, W1,
                                           out_sigma, out_rgb, B);
}
